// ReverseDistributionMnist_55284819034625
// MI455X (gfx1250) — matrix operand order repaired and checked by reading, not compiled, not run on hardware
//
#include <hip/hip_runtime.h>
#include <stdint.h>

typedef __attribute__((ext_vector_type(8)))  _Float16 v8h;
typedef __attribute__((ext_vector_type(16))) _Float16 v16h;
typedef __attribute__((ext_vector_type(8)))  float    v8f;

#define TEMP_INV 10.0f   // 1 / TEMPERATURE

// ------------------------------------------------------------------
// K0: transpose-convert  W[K][N] (f32, row-major) -> Wt[N][K] (f16)
// ------------------------------------------------------------------
__global__ void k_transpose_f16(const float* __restrict__ W,
                                _Float16* __restrict__ Wt, int K, int N) {
  int idx = blockIdx.x * blockDim.x + threadIdx.x;
  if (idx >= K * N) return;
  int k = idx / N, n = idx - k * N;
  Wt[(size_t)n * K + k] = (_Float16)W[idx];
}

// ------------------------------------------------------------------
// K1: h1 = relu(y @ W1 + b1)  (B x 128), K=10 -> scalar FMA, f16 out
// ------------------------------------------------------------------
__global__ void k_layer1(const float* __restrict__ y, const float* __restrict__ W1,
                         const float* __restrict__ b1, _Float16* __restrict__ h1,
                         int B) {
  int idx = blockIdx.x * blockDim.x + threadIdx.x;
  if (idx >= B * 128) return;
  int row = idx >> 7, col = idx & 127;
  float s = b1[col];
#pragma unroll
  for (int i = 0; i < 10; ++i)
    s = fmaf(y[row * 10 + i], W1[i * 128 + col], s);
  h1[idx] = (_Float16)fmaxf(s, 0.0f);
}

// ------------------------------------------------------------------
// WMMA GEMM:  C(B x N) = relu(A(B x K) @ W(K x N) + bias)
// A row-major f16; Wt is W transposed (N x K) row-major f16.
// 1 wave / block; wave computes 16 rows x NT N-tiles. NT is a
// compile-time constant so the K-loop body is branch-free (no
// accumulator duplication across wave-uniform guards).
// ------------------------------------------------------------------
template <int K, int N, int NT, bool OUT_F16>
__global__ __launch_bounds__(32)
void k_gemm_relu(const _Float16* __restrict__ A, const _Float16* __restrict__ Wt,
                 const float* __restrict__ bias, _Float16* __restrict__ Cf16,
                 float* __restrict__ Cf32, int tn_base) {
  const int l   = threadIdx.x;       // 0..31
  const int mlo = l & 15;
  const int hi  = l >> 4;            // 0 or 1
  const int row0 = blockIdx.x * 16;
  const int tn0  = tn_base + blockIdx.y * NT;

  v8f acc[NT];
#pragma unroll
  for (int t = 0; t < NT; ++t)
#pragma unroll
    for (int i = 0; i < 8; ++i) acc[t][i] = 0.0f;

  const _Float16* Arow = A + (size_t)(row0 + mlo) * K;

#pragma unroll 4
  for (int k0 = 0; k0 < K; k0 += 32) {
    // A fragment (16x32 f16): lane<16 holds K = k0+{0..7} and k0+16+{0..7};
    // lane>=16 holds K = k0+8+{0..7} and k0+24+{0..7}   (ISA 7.12.2)
    v8h alo = *(const v8h*)(Arow + k0 + hi * 8);
    v8h ahi = *(const v8h*)(Arow + k0 + 16 + hi * 8);
    v16h a;
#pragma unroll
    for (int i = 0; i < 8; ++i) { a[i] = alo[i]; a[i + 8] = ahi[i]; }

#pragma unroll
    for (int t = 0; t < NT; ++t) {
      // B fragment (32x16 f16): lane holds column n=mlo, K = k0+hi*8+{0..7} and k0+16+hi*8+{0..7}
      const _Float16* Brow = Wt + (size_t)((tn0 + t) * 16 + mlo) * K + k0 + hi * 8;
      v8h blo = *(const v8h*)(Brow);
      v8h bhi = *(const v8h*)(Brow + 16);
      v16h b;
#pragma unroll
      for (int i = 0; i < 8; ++i) { b[i] = blo[i]; b[i + 8] = bhi[i]; }

      acc[t] = __builtin_amdgcn_wmma_f32_16x16x32_f16(
          /*neg_a=*/false, a, /*neg_b=*/false, b,
          /*c_mod=*/(short)0, acc[t], /*reuse_a=*/false, /*reuse_b=*/false);
    }
  }

  // Epilogue: C/D layout -> lane mlo = column, VGPR r -> row r + hi*8
#pragma unroll
  for (int t = 0; t < NT; ++t) {
    const int n = (tn0 + t) * 16 + mlo;
    const float bv = bias[n];
#pragma unroll
    for (int r = 0; r < 8; ++r) {
      const int m = r + hi * 8;
      float v = fmaxf(acc[t][r] + bv, 0.0f);
      size_t o = (size_t)(row0 + m) * N + n;
      if (OUT_F16) Cf16[o] = (_Float16)v;
      else         Cf32[o] = v;
    }
  }
}

// ------------------------------------------------------------------
// Counter-based RNG (SplitMix64 hash) -> uniform in (0,1)
// ------------------------------------------------------------------
__device__ __forceinline__ float u01(uint32_t row, uint32_t col, uint32_t s) {
  uint64_t x = ((uint64_t)row << 40) ^ ((uint64_t)col << 20) ^ (uint64_t)s
             ^ 0x9E3779B97F4A7C15ull;
  x ^= x >> 30; x *= 0xBF58476D1CE4E5B9ull;
  x ^= x >> 27; x *= 0x94D049BB133111EBull;
  x ^= x >> 31;
  uint32_t m = (uint32_t)(x >> 40);               // 24 random bits
  return ((float)m + 0.5f) * (1.0f / 16777216.0f);
}

__device__ __forceinline__ float wg_reduce(float v, float* red, bool is_max) {
  // wave32 reduction
#pragma unroll
  for (int off = 16; off > 0; off >>= 1) {
    float o = __shfl_xor(v, off, 32);
    v = is_max ? fmaxf(v, o) : (v + o);
  }
  const int wid = threadIdx.x >> 5;
  if ((threadIdx.x & 31) == 0) red[wid] = v;
  __syncthreads();
  if (threadIdx.x < 8) {
    float x = red[threadIdx.x];
#pragma unroll
    for (int off = 4; off > 0; off >>= 1) {
      float o = __shfl_xor(x, off, 32);
      x = is_max ? fmaxf(x, o) : (x + o);
    }
    if (threadIdx.x == 0) red[0] = x;
  }
  __syncthreads();
  float r = red[0];
  __syncthreads();
  return r;
}

// ------------------------------------------------------------------
// K4: img = sum_{s<k} softmax( (log p + g_s) / T )   per row of 784
// ------------------------------------------------------------------
__global__ __launch_bounds__(256)
void k_gumbel(const float* __restrict__ proba, const int* __restrict__ kptr,
              float* __restrict__ img) {
  __shared__ float red[8];
  const int row = blockIdx.x;
  const int tid = threadIdx.x;
  const int kk  = kptr[0];

  int   col[4]; bool own[4]; float zb[4], acc[4];
#pragma unroll
  for (int e = 0; e < 4; ++e) {
    col[e] = tid + e * 256;
    own[e] = (col[e] < 784);
    acc[e] = 0.0f;
    zb[e]  = own[e]
           ? TEMP_INV * __logf(fmaxf(proba[(size_t)row * 784 + col[e]], 1e-30f))
           : -3.0e38f;
  }

  for (int s = 0; s < kk; ++s) {
    float z[4], lmax = -3.0e38f;
#pragma unroll
    for (int e = 0; e < 4; ++e) {
      if (own[e]) {
        float u   = u01((uint32_t)row, (uint32_t)col[e], (uint32_t)s);
        float nlu = -__logf(u);                     // -log(u) > 0
        z[e] = zb[e] - TEMP_INV * __logf(nlu);      // (log p + g)/T
        lmax = fmaxf(lmax, z[e]);
      } else {
        z[e] = -3.0e38f;
      }
    }
    float gmax = wg_reduce(lmax, red, true);

    float ez[4], lsum = 0.0f;
#pragma unroll
    for (int e = 0; e < 4; ++e) {
      ez[e] = own[e] ? __expf(z[e] - gmax) : 0.0f;
      lsum += ez[e];
    }
    float gsum = wg_reduce(lsum, red, false);
    float inv  = 1.0f / gsum;
#pragma unroll
    for (int e = 0; e < 4; ++e) acc[e] = fmaf(ez[e], inv, acc[e]);
  }

#pragma unroll
  for (int e = 0; e < 4; ++e)
    if (own[e]) img[(size_t)row * 784 + col[e]] = acc[e];
}

// ------------------------------------------------------------------
extern "C" void kernel_launch(void* const* d_in, const int* in_sizes, int n_in,
                              void* d_out, int out_size, void* d_ws, size_t ws_size,
                              hipStream_t stream) {
  const float* y  = (const float*)d_in[0];
  const float* W1 = (const float*)d_in[1];
  const float* b1 = (const float*)d_in[2];
  const float* W2 = (const float*)d_in[3];
  const float* b2 = (const float*)d_in[4];
  const float* W3 = (const float*)d_in[5];
  const float* b3 = (const float*)d_in[6];
  const int*   kp = (const int*)d_in[7];

  const int B = in_sizes[0] / 10;                  // 16384

  // workspace layout (16B-aligned slices)
  char* ws = (char*)d_ws;
  _Float16* h1f16 = (_Float16*)(ws);                                   // B*128*2
  _Float16* W2t   = (_Float16*)(ws + (size_t)B * 128 * 2);             // 512*128*2
  _Float16* W3t   = (_Float16*)(ws + (size_t)B * 128 * 2 + 512 * 128 * 2);
  _Float16* h2f16 = (_Float16*)(ws + (size_t)B * 128 * 2 + 512 * 128 * 2
                                   + (size_t)784 * 512 * 2);           // B*512*2

  float* img   = (float*)d_out;
  float* proba = (float*)d_out + (size_t)B * 784;

  // K0: weight transpose-converts
  {
    int n2 = 128 * 512;
    k_transpose_f16<<<(n2 + 255) / 256, 256, 0, stream>>>(W2, W2t, 128, 512);
    int n3 = 512 * 784;
    k_transpose_f16<<<(n3 + 255) / 256, 256, 0, stream>>>(W3, W3t, 512, 784);
  }

  // K1: layer 1 (K=10, scalar)
  {
    int n = B * 128;
    k_layer1<<<(n + 255) / 256, 256, 0, stream>>>(y, W1, b1, h1f16, B);
  }

  // K2: layer 2 WMMA  h2 = relu(h1 @ W2 + b2)   (N=512 -> 8 chunks of 64)
  {
    dim3 grid(B / 16, 512 / 64);
    k_gemm_relu<128, 512, 4, true><<<grid, 32, 0, stream>>>(
        h1f16, W2t, b2, h2f16, nullptr, 0);
  }

  // K3: layer 3 WMMA  proba = relu(h2 @ W3 + b3)
  //     N=784 = 49 tiles: 12 chunks of 4 tiles + 1 single-tile launch
  {
    dim3 grid(B / 16, 12);
    k_gemm_relu<512, 784, 4, false><<<grid, 32, 0, stream>>>(
        h2f16, W3t, b3, nullptr, proba, 0);
    dim3 grid1(B / 16, 1);
    k_gemm_relu<512, 784, 1, false><<<grid1, 32, 0, stream>>>(
        h2f16, W3t, b3, nullptr, proba, 48);
  }

  // K4: k-step Gumbel-softmax accumulation
  k_gumbel<<<B, 256, 0, stream>>>(proba, kp, img);
}
